// GCN_52304111731095
// MI455X (gfx1250) — matrix operand order repaired and checked by reading, not compiled, not run on hardware
//
#include <hip/hip_runtime.h>
#include <hip/hip_bf16.h>

// GCN on MI455X (gfx1250):
//   h = x@fc1_W + b;  3x: h = relu(adj1@(h@W_h[i]) + b_h[i] + h);
//   out = relu(adj1@(h@W_out) + b_out)
// Strategy: convert adj1 (256MB fp32) once to f16 (128MB -> resident in 192MB
// L2 across all 4 adjacency passes; fp32 source is read with NT hints so it
// doesn't thrash L2). Every matmul runs on v_wmma_f32_16x16x32_f16 with fp32
// accumulation. B operands are pre-packed into exact WMMA B-fragment order so
// each lane issues one coalesced 32B load per fragment. Each workgroup = 2
// waves splitting the K range, combined once through LDS (doubles occupancy
// with zero extra global traffic).

typedef __attribute__((ext_vector_type(16))) _Float16 v16h;
typedef __attribute__((ext_vector_type(8)))  _Float16 v8h;
typedef __attribute__((ext_vector_type(4)))  _Float16 v4h;
typedef __attribute__((ext_vector_type(8)))  float    v8f;
typedef __attribute__((ext_vector_type(4)))  float    v4f;

// ---------------------------------------------------------------------------
// fp32 -> fp16 elementwise conversion, 4 elements/thread.
// Source is read non-temporally: the 256MB fp32 adjacency is dead after this
// pass and must not evict the f16 working set from L2.
// ---------------------------------------------------------------------------
__global__ __launch_bounds__(256) void cvt_f32_f16_kernel(
    const float* __restrict__ in, _Float16* __restrict__ out, size_t n4)
{
    size_t i = (size_t)blockIdx.x * blockDim.x + threadIdx.x;
    if (i >= n4) return;
    v4f v = __builtin_nontemporal_load((const v4f*)(in + i * 4));
    v4h h;
    h.x = (_Float16)v.x; h.y = (_Float16)v.y;
    h.z = (_Float16)v.z; h.w = (_Float16)v.w;
    *(v4h*)(out + i * 4) = h;
}

// ---------------------------------------------------------------------------
// Pack a row-major fp32 [K x ncols] matrix (leading dim ldt) into f16 WMMA
// B-fragment order for 16x16x32 f16 WMMA:
//   Bp[kb][nt][lane][j]  (j = 0..15 halves, contiguous 32B per lane)
//   element = T[kb*32 + (lane>>4)*8 + (j&7) + (j>>3)*16][nt*16 + (lane&15)]   (0 if col OOB)
// ---------------------------------------------------------------------------
__global__ __launch_bounds__(256) void pack_b_kernel(
    const float* __restrict__ T, int ldt, int ncols, int K, int NT,
    _Float16* __restrict__ Bp)
{
    size_t idx = (size_t)blockIdx.x * blockDim.x + threadIdx.x;
    size_t total = (size_t)(K >> 5) * (size_t)NT * 512u;
    if (idx >= total) return;
    int j    = (int)(idx & 15);
    int lane = (int)((idx >> 4) & 31);
    int nt   = (int)((idx >> 9) % (size_t)NT);
    size_t kb = idx / ((size_t)NT * 512u);
    int k = (int)(kb * 32) + (lane >> 4) * 8 + (j & 7) + (j >> 3) * 16;
    int n = nt * 16 + (lane & 15);
    float v = (n < ncols) ? T[(size_t)k * ldt + n] : 0.0f;
    Bp[idx] = (_Float16)v;
}

// ---------------------------------------------------------------------------
// WMMA GEMM: C[16*gridDim.x x NT*16] = A(f16,[M x K] row-major) @ Bp(packed)
// Workgroup = 64 threads = 2 waves; wave w accumulates k-blocks
// [w*K/2, (w+1)*K/2). Wave 1 dumps its accumulators to LDS; after the
// barrier wave 0 combines and runs the fused epilogue:
//   + bias[n], + residual[m][n], ReLU, fp32 store, fp16 store.
// All branches are wave-uniform => EXEC is all-ones at every WMMA.
// Requires K % 64 == 0 (K = 8192 / 1024 / 64 here).
// ---------------------------------------------------------------------------
template <int NT>
__global__ __launch_bounds__(64) void gemm_wmma_kernel(
    const _Float16* __restrict__ A, int lda,
    const _Float16* __restrict__ Bp, int K,
    const float* __restrict__ bias,
    const float* __restrict__ res, int ldres,
    float* __restrict__ Cf32, int ldc, int relu, int ncols,
    _Float16* __restrict__ Cf16, int ldh)
{
    __shared__ v8f lsum[NT * 32];

    const int lane  = threadIdx.x & 31;     // wave32 lane
    const int wave  = threadIdx.x >> 5;     // 0 or 1: K-split
    const int mtile = blockIdx.x;
    const int mrow  = mtile * 16 + (lane & 15);
    const int kbase = (lane >> 4) * 8;      // A-frag: lanes 16-31 start at K=8

    v8f acc[NT] = {};

    const int nkb_half = (K >> 5) >> 1;     // k-blocks per wave
    const int kb0      = wave * nkb_half;

    // A fragment layout (16-bit A 16x32): per lane two contiguous 16B chunks:
    //   halves 0..7  = A[m][kb*32 + kbase + 0..7]
    //   halves 8..15 = A[m][kb*32 + kbase + 16..23]
    const _Float16* arow = A + (size_t)mrow * lda + (size_t)kb0 * 32 + kbase;
    const v16h* bbase = (const v16h*)Bp + (size_t)kb0 * NT * 32 + lane;
    for (int kb = 0; kb < nkb_half; ++kb) {
        union { v16h v; v8h h[2]; } a;
        a.h[0] = *(const v8h*)(arow);
        a.h[1] = *(const v8h*)(arow + 16);
        __builtin_prefetch(arow + 128, 0, 3);   // near-scope prefetch 4 tiles ahead
        arow += 32;
        const v16h* bb = bbase + (size_t)kb * NT * 32;
#pragma unroll
        for (int t = 0; t < NT; ++t) {
            v16h b = bb[t * 32];
            acc[t] = __builtin_amdgcn_wmma_f32_16x16x32_f16(
                false, a.v, false, b, (short)0, acc[t], false, false);
        }
    }

    // Combine the two K-halves through LDS.
    if (wave == 1) {
#pragma unroll
        for (int t = 0; t < NT; ++t) lsum[t * 32 + lane] = acc[t];
    }
    __syncthreads();
    if (wave == 0) {
#pragma unroll
        for (int t = 0; t < NT; ++t) acc[t] += lsum[t * 32 + lane];

        // C/D layout: VGPR r, lanes 0-15 -> M=r, lanes 16-31 -> M=8+r; N=lane&15
        const int chalf = lane >> 4;
        const int ncol  = lane & 15;
#pragma unroll
        for (int t = 0; t < NT; ++t) {
            const int n = t * 16 + ncol;
            const bool nok = (n < ncols);
            const float badd = (bias != nullptr && nok) ? bias[n] : 0.0f;
#pragma unroll
            for (int r = 0; r < 8; ++r) {
                const int m = mtile * 16 + chalf * 8 + r;
                float v = acc[t][r] + badd;
                if (res)  v += res[(size_t)m * ldres + n];
                if (relu) v = fmaxf(v, 0.0f);
                if (Cf32 && nok) Cf32[(size_t)m * ldc + n] = v;
                if (Cf16 && nok) Cf16[(size_t)m * ldh + n] = (_Float16)v;
            }
        }
    }
}

// ---------------------------------------------------------------------------
extern "C" void kernel_launch(void* const* d_in, const int* in_sizes, int n_in,
                              void* d_out, int out_size, void* d_ws, size_t ws_size,
                              hipStream_t stream) {
    (void)in_sizes; (void)n_in; (void)out_size; (void)ws_size;
    constexpr int N = 8192, IN_DIM = 1024, HID = 64, NCLS = 40;

    const float* x     = (const float*)d_in[0];
    const float* adj1  = (const float*)d_in[1];
    // d_in[2] = adj2 (unused by the reference computation)
    const float* fc1_W = (const float*)d_in[3];
    const float* fc1_b = (const float*)d_in[4];
    const float* W_h   = (const float*)d_in[5];   // [3,64,64]
    const float* b_h   = (const float*)d_in[6];   // [3,64]
    const float* W_out = (const float*)d_in[7];   // [64,40]
    const float* b_out = (const float*)d_in[8];   // [40]
    float* out = (float*)d_out;                   // [8192,40]

    // Workspace carve-out (~150 MB): adj f16 (128MB, L2-resident) + x f16 + small bufs
    char* ws = (char*)d_ws;
    size_t off = 0;
    auto carve = [&](size_t bytes) -> char* {
        char* p = ws + off;
        off += (bytes + 255) & ~(size_t)255;
        return p;
    };
    _Float16* adjh  = (_Float16*)carve((size_t)N * N * 2);        // 128 MB
    _Float16* xh    = (_Float16*)carve((size_t)N * IN_DIM * 2);   // 16 MB
    float*    Hf32  = (float*)   carve((size_t)N * HID * 4);      // 2 MB
    _Float16* Hf16  = (_Float16*)carve((size_t)N * HID * 2);      // 1 MB
    float*    Tf32  = (float*)   carve((size_t)N * HID * 4);      // 2 MB
    _Float16* BpBig = (_Float16*)carve((size_t)(N / 32) * 4 * 512 * 2);      // 1 MB
    _Float16* BpW   = (_Float16*)carve((size_t)(IN_DIM / 32) * 4 * 512 * 2); // 128 KB

    const int MT = N / 16;  // 512 M-tiles

    auto grid_for = [](size_t elems) { return (unsigned)((elems + 255) / 256); };

    // 1) adj1 fp32 -> f16 (once; f16 copy stays hot in 192MB L2 for 4 passes)
    cvt_f32_f16_kernel<<<grid_for((size_t)N * N / 4), 256, 0, stream>>>(
        adj1, adjh, (size_t)N * N / 4);
    // 2) x fp32 -> f16
    cvt_f32_f16_kernel<<<grid_for((size_t)N * IN_DIM / 4), 256, 0, stream>>>(
        x, xh, (size_t)N * IN_DIM / 4);

    // 3) fc1: H = x @ fc1_W + b   (K=1024, NT=4)
    pack_b_kernel<<<grid_for((size_t)(IN_DIM / 32) * 4 * 512), 256, 0, stream>>>(
        fc1_W, HID, HID, IN_DIM, 4, BpW);
    gemm_wmma_kernel<4><<<MT, 64, 0, stream>>>(
        xh, IN_DIM, BpW, IN_DIM,
        fc1_b, nullptr, 0, Hf32, HID, /*relu=*/0, HID, Hf16, HID);

    // 4) hidden GraphConvolution layers
    for (int i = 0; i < 3; ++i) {
        // T = H @ W_h[i]  (K=64)
        pack_b_kernel<<<grid_for((size_t)(HID / 32) * 4 * 512), 256, 0, stream>>>(
            W_h + (size_t)i * HID * HID, HID, HID, HID, 4, BpW);
        gemm_wmma_kernel<4><<<MT, 64, 0, stream>>>(
            Hf16, HID, BpW, HID,
            nullptr, nullptr, 0, Tf32, HID, /*relu=*/0, HID, nullptr, 0);
        // pack T for the adjacency pass
        pack_b_kernel<<<grid_for((size_t)(N / 32) * 4 * 512), 256, 0, stream>>>(
            Tf32, HID, HID, N, 4, BpBig);
        // H = relu(adj1 @ T + b_h[i] + H)   (K=8192; in-place per-element update)
        gemm_wmma_kernel<4><<<MT, 64, 0, stream>>>(
            adjh, N, BpBig, N,
            b_h + (size_t)i * HID, Hf32, HID, Hf32, HID, /*relu=*/1, HID, Hf16, HID);
    }

    // 5) output layer (N_CLASS=40 -> 3 n-tiles, padded to 48 with zeros)
    pack_b_kernel<<<grid_for((size_t)(HID / 32) * 3 * 512), 256, 0, stream>>>(
        W_out, NCLS, NCLS, HID, 3, BpW);
    gemm_wmma_kernel<3><<<MT, 64, 0, stream>>>(
        Hf16, HID, BpW, HID,
        nullptr, nullptr, 0, Tf32, 48, /*relu=*/0, 48, nullptr, 0);
    pack_b_kernel<<<grid_for((size_t)(N / 32) * 3 * 512), 256, 0, stream>>>(
        Tf32, 48, 48, N, 3, BpBig);
    // out = relu(adj1 @ T + b_out), store only the first 40 columns
    gemm_wmma_kernel<3><<<MT, 64, 0, stream>>>(
        adjh, N, BpBig, N,
        b_out, nullptr, 0, out, NCLS, /*relu=*/1, NCLS, nullptr, 0);
}
